// RotaryAttention_23295902614259
// MI455X (gfx1250) — matrix operand order repaired and checked by reading, not compiled, not run on hardware
//
#include <hip/hip_runtime.h>

typedef __attribute__((ext_vector_type(16))) _Float16 v16h;
typedef __attribute__((ext_vector_type(8)))  _Float16 v8h;
typedef __attribute__((ext_vector_type(4)))  _Float16 v4h;
typedef __attribute__((ext_vector_type(8)))  float    v8f;
typedef __attribute__((ext_vector_type(4)))  float    v4f;
typedef __attribute__((ext_vector_type(4)))  unsigned int u32x4;
typedef __attribute__((ext_vector_type(8)))  unsigned int u32x8;

#define DEVINL __device__ __forceinline__

DEVINL v8h cvt_v8(v4f a, v4f b) {
  v8h r;
  r[0] = (_Float16)a[0]; r[1] = (_Float16)a[1]; r[2] = (_Float16)a[2]; r[3] = (_Float16)a[3];
  r[4] = (_Float16)b[0]; r[5] = (_Float16)b[1]; r[6] = (_Float16)b[2]; r[7] = (_Float16)b[3];
  return r;
}

DEVINL v16h cat16(v8h lo, v8h hi) {
  return __builtin_shufflevector(lo, hi, 0,1,2,3,4,5,6,7,8,9,10,11,12,13,14,15);
}

DEVINL v8f wmma16(v16h a, v16h b, v8f c) {
  // D = A(16x32 f16) * B(32x16 f16) + C(16x16 f32)
  return __builtin_amdgcn_wmma_f32_16x16x32_f16(
      /*neg_a=*/false, a, /*neg_b=*/false, b,
      /*c_mod=*/(short)0, c, /*reuse_a=*/false, /*reuse_b=*/false);
}

// ---------------------------------------------------------------------------
// GEMM: C[M,N] = A[M,K] (f32, row-major) @ B[K,N] (f32, row-major) + bias[N]
// BM=128 BN=128 BK=32, 256 threads (8 waves), wave = 64x32 sub-tile (4x2 WMMA)
// ---------------------------------------------------------------------------
template <bool OUT_F16>
__global__ __launch_bounds__(256)
void gemm_bias_kernel(const float* __restrict__ A, const float* __restrict__ B,
                      const float* __restrict__ bias, void* __restrict__ Cv,
                      int M, int N, int K) {
  constexpr int BK = 32, LDA = BK + 8, LDB = BK + 8;
  __shared__ __align__(16) _Float16 As[128 * LDA];
  __shared__ __align__(16) _Float16 Bs[128 * LDB];   // transposed: [n][k]

  const int tid  = threadIdx.x;
  const int lane = tid & 31, wave = tid >> 5;
  const int lrow = lane & 15, hl = lane >> 4;
  const int wm = wave >> 2, wn = wave & 3;           // 2 x 4 wave grid
  const int m0 = blockIdx.y * 128, n0 = blockIdx.x * 128;

  const int ar = tid >> 1, ac = (tid & 1) * 16;      // A tile: 128 x 32
  const int br = tid >> 3, bc = (tid & 7) * 16;      // B tile: 32 x 128

  v8f acc[4][2];
#pragma unroll
  for (int i = 0; i < 4; i++)
#pragma unroll
    for (int j = 0; j < 2; j++) { v8f z = {}; acc[i][j] = z; }

  for (int k0 = 0; k0 < K; k0 += BK) {
    __syncthreads();
    {
      const float* ap = A + (size_t)(m0 + ar) * K + (k0 + ac);
      v4f a0 = ((const v4f*)ap)[0], a1 = ((const v4f*)ap)[1];
      v4f a2 = ((const v4f*)ap)[2], a3 = ((const v4f*)ap)[3];
      v8h* as = (v8h*)&As[ar * LDA + ac];
      as[0] = cvt_v8(a0, a1);
      as[1] = cvt_v8(a2, a3);

      const float* bp = B + (size_t)(k0 + br) * N + (n0 + bc);
      v4f b0 = ((const v4f*)bp)[0], b1 = ((const v4f*)bp)[1];
      v4f b2 = ((const v4f*)bp)[2], b3 = ((const v4f*)bp)[3];
#pragma unroll
      for (int i = 0; i < 4; i++) {
        Bs[(bc +      i) * LDB + br] = (_Float16)b0[i];
        Bs[(bc +  4 + i) * LDB + br] = (_Float16)b1[i];
        Bs[(bc +  8 + i) * LDB + br] = (_Float16)b2[i];
        Bs[(bc + 12 + i) * LDB + br] = (_Float16)b3[i];
      }
      if (k0 + BK < K) {
        __builtin_prefetch(A + (size_t)(m0 + ar) * K + (k0 + BK + ac), 0, 1);
        __builtin_prefetch(B + (size_t)(k0 + BK + br) * N + (n0 + bc), 0, 1);
      }
    }
    __syncthreads();

    const int ka = hl * 8;
    v16h af[4];
#pragma unroll
    for (int i = 0; i < 4; i++) {
      const _Float16* p = &As[(wm * 64 + i * 16 + lrow) * LDA];
      af[i] = cat16(*(const v8h*)(p + ka), *(const v8h*)(p + 16 + ka));
    }
    v16h bf[2];
#pragma unroll
    for (int j = 0; j < 2; j++) {
      const _Float16* p = &Bs[(wn * 32 + j * 16 + lrow) * LDB] + hl * 8;
      bf[j] = cat16(*(const v8h*)p, *(const v8h*)(p + 16));
    }
#pragma unroll
    for (int i = 0; i < 4; i++)
#pragma unroll
      for (int j = 0; j < 2; j++)
        acc[i][j] = wmma16(af[i], bf[j], acc[i][j]);
  }

#pragma unroll
  for (int j = 0; j < 2; j++) {
    const int n = n0 + wn * 32 + j * 16 + lrow;
    const float bv = bias[n];
#pragma unroll
    for (int i = 0; i < 4; i++) {
      const int mr = m0 + wm * 64 + i * 16 + hl * 8;
#pragma unroll
      for (int r = 0; r < 8; r++) {
        const float val = acc[i][j][r] + bv;
        if constexpr (OUT_F16)
          ((_Float16*)Cv)[(size_t)(mr + r) * N + n] = (_Float16)val;
        else
          ((float*)Cv)[(size_t)(mr + r) * N + n] = val;
      }
    }
  }
}

// ---------------------------------------------------------------------------
// RMSNorm + RoPE, one wave per 128-wide head row, f32 in -> f16 out
// ---------------------------------------------------------------------------
__global__ __launch_bounds__(256)
void rmsnorm_rope_kernel(const float* __restrict__ in, const float* __restrict__ nscale,
                         _Float16* __restrict__ out, int heads, int seq, float oscale) {
  const int wave = threadIdx.x >> 5, lane = threadIdx.x & 31;
  const size_t row = (size_t)blockIdx.x * 8 + wave;
  const int pos = (int)((row / heads) % seq);

  v4f x = *(const v4f*)(in + row * 128 + lane * 4);
  float ss = x[0]*x[0] + x[1]*x[1] + x[2]*x[2] + x[3]*x[3];
#pragma unroll
  for (int off = 16; off > 0; off >>= 1) ss += __shfl_xor(ss, off, 32);
  const float inv = rsqrtf(ss * (1.0f / 128.0f) + 1e-6f);

  v4f sc = *(const v4f*)(nscale + lane * 4);
  const float a0 = x[0] * inv * sc[0], a1 = x[1] * inv * sc[1];
  const float a2 = x[2] * inv * sc[2], a3 = x[3] * inv * sc[3];

  const int p0 = lane * 2, p1 = lane * 2 + 1;
  const float kln = 0.14391156f;                 // ln(10000)/64
  float s0, c0, s1, c1;
  __sincosf((float)pos * __expf(-(float)p0 * kln), &s0, &c0);
  __sincosf((float)pos * __expf(-(float)p1 * kln), &s1, &c1);

  v4h r;
  r[0] = (_Float16)((a0 * c0 - a1 * s0) * oscale);
  r[1] = (_Float16)((a0 * s0 + a1 * c0) * oscale);
  r[2] = (_Float16)((a2 * c1 - a3 * s1) * oscale);
  r[3] = (_Float16)((a2 * s1 + a3 * c1) * oscale);
  *(v4h*)(out + row * 128 + lane * 4) = r;
}

// ---------------------------------------------------------------------------
// Flash attention, 1 WG = (128 q-rows, 1 head, 1 batch), 8 waves x 16 rows
// Q layout [b,s,H*128] f16 (rope'd, * hd^-0.5); K/V layout [b,s,G*128] f16
// K tile is pulled in by the Tensor Data Mover (wave 0) with LDS row padding;
// V tile is transposed into LDS by all waves concurrently.
// S tiles are retired pairwise (raw S -> LDS); exp folds into the PV A-frag.
// ---------------------------------------------------------------------------
__global__ __launch_bounds__(256)
void attention_kernel(const _Float16* __restrict__ Q, const _Float16* __restrict__ Kc,
                      const _Float16* __restrict__ Vc, float* __restrict__ O, int seq) {
  constexpr int LDK = 128 + 8, LDV = 128 + 8, LDP = 128 + 8;
  __shared__ __align__(16) _Float16 Ks[128 * LDK];       // [key][d]   (TDM-filled)
  __shared__ __align__(16) _Float16 Vt[128 * LDV];       // [d][key]   (transposed)
  __shared__ __align__(16) _Float16 Ps[8 * 16 * LDP];    // per-wave raw-S staging
  __shared__ __align__(16) float NM[8][16];              // per-wave row max
  __shared__ __align__(16) float AL[8][16];              // per-wave row alpha
  __shared__ __align__(16) float LL[8][16];              // per-wave row l-sum

  const int tid = threadIdx.x, lane = tid & 31, wave = tid >> 5;
  const int lrow = lane & 15, hl = lane >> 4;
  const int qb = blockIdx.x, h = blockIdx.y, bz = blockIdx.z;
  const int g = h >> 2;                                   // P = H/G = 4
  const float scl = 0.08838834764831845f;                 // hd^-0.5 (2nd factor)
  const size_t qbase  = ((size_t)bz * seq) * 2048 + (size_t)h * 128;
  const size_t kvbase = ((size_t)bz * seq) * 512  + (size_t)g * 128;

  v16h qf[4];
  {
    const _Float16* qp = Q + qbase + (size_t)(qb * 128 + wave * 16 + lrow) * 2048;
#pragma unroll
    for (int c = 0; c < 4; c++)
      qf[c] = cat16(*(const v8h*)(qp + c * 32 + hl * 8),
                    *(const v8h*)(qp + c * 32 + 16 + hl * 8));
  }

  v8f accO[8];
#pragma unroll
  for (int t = 0; t < 8; t++) { v8f z = {}; accO[t] = z; }
  float mrun_a = -1e30f, lrun_a = 0.f;                    // A-layout: row = lrow
  const int rowoff = hl * 8;
  const int qrow0 = qb * 128 + wave * 16 + rowoff;
  _Float16* pw = &Ps[wave * 16 * LDP];

  for (int kb = 0; kb <= qb; ++kb) {
    __syncthreads();
    if (wave == 0) {
      // --- TDM: DMA the 128x128 f16 K tile into LDS with 16B row padding ---
      const unsigned long long ga =
          (unsigned long long)(const void*)(Kc + kvbase + (size_t)(kb * 128) * 512);
      const unsigned int lds = (unsigned int)(unsigned long long)(const void*)Ks;
      u32x4 g0;
      g0[0] = 1u;                                            // count=1 (valid user D#)
      g0[1] = lds;                                           // lds_addr
      g0[2] = (unsigned int)ga;                              // global_addr[31:0]
      g0[3] = (unsigned int)((ga >> 32) & 0x01FFFFFFu)       // global_addr[56:32]
              | (2u << 30);                                  // type=2 (image)
      u32x8 g1;
      g1[0] = (1u << 16)      // data_size = 1 -> 2 bytes
            | (1u << 20)      // pad_enable
            | (5u << 22)      // pad_interval = 5 -> every 64 DWORDs (256B = 1 row)
            | (3u << 25);     // pad_amount   = 3 -> 4 DWORDs (16B) => LDK stride 136
      g1[1] = (512u << 16);   // atomic_barrier=0; tensor_dim0[15:0] = 512
      g1[2] = 0u;             // tensor_dim0[31:16]=0; tensor_dim1[15:0]=0
      g1[3] = 1u              // tensor_dim1[31:16]=1 -> tensor_dim1 = 65536
            | (128u << 16);   // tile_dim0 = 128 elements
      g1[4] = 128u;           // tile_dim1 = 128 rows, tile_dim2 = 0
      g1[5] = 512u;           // tensor_dim0_stride[31:0] = 512 elements
      g1[6] = 0u;             // stride0 hi, stride1 lo
      g1[7] = 0u;             // stride1 hi
      u32x4 g2 = {}, g3 = {}; // groups 2/3: tile_dim3/4 = 0 (unused)
      asm volatile("tensor_load_to_lds %0, %1, %2, %3"
                   :: "s"(g0), "s"(g1), "s"(g2), "s"(g3) : "memory");
      __builtin_amdgcn_s_wait_tensorcnt(0);
    }
    {
      // --- all waves: V tile transposed into LDS (overlaps the TDM copy) ---
      const int j = tid >> 1, d0 = (tid & 1) * 64;
      const _Float16* vp = Vc + kvbase + (size_t)(kb * 128 + j) * 512 + d0;
#pragma unroll
      for (int u = 0; u < 8; u++) {
        v8h vv = ((const v8h*)vp)[u];
#pragma unroll
        for (int e = 0; e < 8; e++) Vt[(d0 + u * 8 + e) * LDV + j] = vv[e];
      }
    }
    __syncthreads();

    // ---- S = Q @ K^T, tiles retired in pairs (raw scaled/masked S -> LDS) --
    const bool diag = (kb == qb);
    float nmax[8];
#pragma unroll
    for (int r = 0; r < 8; r++) nmax[r] = -1e30f;
#pragma unroll
    for (int nt = 0; nt < 8; nt += 2) {
      v8f s0 = {}, s1 = {};
#pragma unroll
      for (int c = 0; c < 4; c++) {
        const _Float16* p0 = &Ks[((nt    ) * 16 + lrow) * LDK + c * 32 + hl * 8];
        const _Float16* p1 = &Ks[((nt + 1) * 16 + lrow) * LDK + c * 32 + hl * 8];
        s0 = wmma16(qf[c], cat16(*(const v8h*)p0, *(const v8h*)(p0 + 16)), s0);
        s1 = wmma16(qf[c], cat16(*(const v8h*)p1, *(const v8h*)(p1 + 16)), s1);
      }
#pragma unroll
      for (int r = 0; r < 8; r++) { s0[r] *= scl; s1[r] *= scl; }
      if (diag) {
        const int kj0 = kb * 128 + (nt    ) * 16 + lrow;
        const int kj1 = kb * 128 + (nt + 1) * 16 + lrow;
#pragma unroll
        for (int r = 0; r < 8; r++) {
          if (kj0 > qrow0 + r) s0[r] = -1e30f;
          if (kj1 > qrow0 + r) s1[r] = -1e30f;
        }
      }
#pragma unroll
      for (int r = 0; r < 8; r++) {
        nmax[r] = fmaxf(nmax[r], fmaxf(s0[r], s1[r]));
        pw[(rowoff + r) * LDP + (nt    ) * 16 + lrow] = (_Float16)s0[r];
        pw[(rowoff + r) * LDP + (nt + 1) * 16 + lrow] = (_Float16)s1[r];
      }
    }

    // ---- row max: C-layout reduce, publish to A-layout via per-wave LDS ----
#pragma unroll
    for (int r = 0; r < 8; r++)
#pragma unroll
      for (int off = 8; off > 0; off >>= 1)
        nmax[r] = fmaxf(nmax[r], __shfl_xor(nmax[r], off, 32));
    if (lrow == 0) {                      // lanes 0 and 16
#pragma unroll
      for (int r = 0; r < 8; r++) NM[wave][rowoff + r] = nmax[r];
    }
    const float nmax_a = NM[wave][lrow];
    const float mnew = fmaxf(mrun_a, nmax_a);
    const float al_a = __expf(mrun_a - mnew);
    mrun_a = mnew;
    if (lane < 16) AL[wave][lrow] = al_a;
    v4f alc0 = ((const v4f*)&AL[wave][rowoff])[0];
    v4f alc1 = ((const v4f*)&AL[wave][rowoff])[1];
#pragma unroll
    for (int nt = 0; nt < 8; nt++)
#pragma unroll
      for (int r = 0; r < 8; r++) accO[nt][r] *= (r < 4) ? alc0[r & 3] : alc1[r & 3];

    // ---- O += P @ V: exp folds into the A-fragment reload ------------------
    float lad = 0.f;
#pragma unroll
    for (int c = 0; c < 4; c++) {
      const _Float16* pa = &pw[lrow * LDP + c * 32];
      v8h lo = *(const v8h*)(pa + hl * 8);
      v8h hi = *(const v8h*)(pa + 16 + hl * 8);
      v16h af;
#pragma unroll
      for (int e = 0; e < 8; e++) {
        const float t0 = __expf((float)lo[e] - mrun_a);
        const float t1 = __expf((float)hi[e] - mrun_a);
        lad += t0 + t1;
        af[e] = (_Float16)t0;
        af[8 + e] = (_Float16)t1;
      }
#pragma unroll
      for (int nt = 0; nt < 8; nt++) {
        const _Float16* pb = &Vt[(nt * 16 + lrow) * LDV + c * 32 + hl * 8];
        accO[nt] = wmma16(af, cat16(*(const v8h*)pb, *(const v8h*)(pb + 16)), accO[nt]);
      }
    }
    lad += __shfl_xor(lad, 16, 32);       // lanes L and L^16 hold the same row
    lrun_a = lrun_a * al_a + lad;
  }

  // ---- normalize and store -------------------------------------------------
  if (lane < 16) LL[wave][lrow] = lrun_a;
  v4f ll0 = ((const v4f*)&LL[wave][rowoff])[0];
  v4f ll1 = ((const v4f*)&LL[wave][rowoff])[1];
  float linv[8];
#pragma unroll
  for (int r = 0; r < 8; r++) linv[r] = 1.0f / ((r < 4) ? ll0[r & 3] : ll1[r & 3]);
#pragma unroll
  for (int nt = 0; nt < 8; nt++) {
    const int n = nt * 16 + lrow;
#pragma unroll
    for (int r = 0; r < 8; r++)
      O[qbase + (size_t)(qrow0 + r) * 2048 + n] = accO[nt][r] * linv[r];
  }
}

// ---------------------------------------------------------------------------
extern "C" void kernel_launch(void* const* d_in, const int* in_sizes, int n_in,
                              void* d_out, int out_size, void* d_ws, size_t ws_size,
                              hipStream_t stream) {
  (void)in_sizes; (void)n_in; (void)out_size; (void)ws_size;
  const float* x   = (const float*)d_in[0];
  const float* Wq  = (const float*)d_in[1];
  const float* bq  = (const float*)d_in[2];
  const float* Wk  = (const float*)d_in[3];
  const float* bk  = (const float*)d_in[4];
  const float* Wv  = (const float*)d_in[5];
  const float* bv  = (const float*)d_in[6];
  const float* Wo  = (const float*)d_in[7];
  const float* bo  = (const float*)d_in[8];
  const float* qns = (const float*)d_in[9];
  const float* kns = (const float*)d_in[10];

  const int b = 16, s = 512, d = 2048, H = 16, G = 4;
  const int M = b * s;                      // 8192
  const float inv_sqrt_hd = 0.08838834764831845f;

  char* ws = (char*)d_ws;
  float*    qf32 = (float*)ws;    ws += (size_t)M * d * 4;     // 64 MiB
  float*    kf32 = (float*)ws;    ws += (size_t)M * 512 * 4;   // 16 MiB
  _Float16* qh   = (_Float16*)ws; ws += (size_t)M * d * 2;     // 32 MiB
  _Float16* kh   = (_Float16*)ws; ws += (size_t)M * 512 * 2;   //  8 MiB
  _Float16* vh   = (_Float16*)ws; ws += (size_t)M * 512 * 2;   //  8 MiB
  float*    aout = (float*)ws;    ws += (size_t)M * d * 4;     // 64 MiB

  dim3 blk(256);
  // Projections: Q (f32 out), K (f32 out), V (f16 out, no norm needed)
  gemm_bias_kernel<false><<<dim3(d / 128, M / 128), blk, 0, stream>>>(x, Wq, bq, qf32, M, d, d);
  gemm_bias_kernel<false><<<dim3(512 / 128, M / 128), blk, 0, stream>>>(x, Wk, bk, kf32, M, 512, d);
  gemm_bias_kernel<true ><<<dim3(512 / 128, M / 128), blk, 0, stream>>>(x, Wv, bv, vh, M, 512, d);

  // RMSNorm + RoPE (q also gets the first hd^-0.5)
  rmsnorm_rope_kernel<<<(M * H) / 8, blk, 0, stream>>>(qf32, qns, qh, H, s, inv_sqrt_hd);
  rmsnorm_rope_kernel<<<(M * G) / 8, blk, 0, stream>>>(kf32, kns, kh, G, s, 1.0f);

  // Flash attention: grid = (s/128 q-blocks, H heads, b batches)
  attention_kernel<<<dim3(s / 128, H, b), blk, 0, stream>>>(qh, kh, vh, aout, s);

  // Output projection
  gemm_bias_kernel<false><<<dim3(d / 128, M / 128), blk, 0, stream>>>(aout, Wo, bo, (float*)d_out, M, d, d);
}
